// MemoryPlusLayer_63934883169083
// MI455X (gfx1250) — matrix operand order repaired and checked by reading, not compiled, not run on hardware
//
#include <hip/hip_runtime.h>
#include <hip/hip_bf16.h>

typedef __bf16 bf16;
typedef __attribute__((ext_vector_type(8)))  __bf16 v8bf;
typedef __attribute__((ext_vector_type(16))) __bf16 v16bf;
typedef __attribute__((ext_vector_type(8)))  float  v8f;

#define NTOK 4096      // B*S
#define DMODEL 1024
#define HDIM 4096      // 4*D
#define SKDIM 256
#define NSUB 512
#define TOPK 32
#define NEG_INF (-3.4e38f)

// ---------------------------------------------------------------------------
// Fragment loads for v_wmma_f32_16x16x32_bf16 (wave32 layouts, CDNA5 ISA 7.12.2)
// A: 16x32 (MxK). lanes 0-15: M=lane, K = k0+0..7 (v0-3), k0+16..23 (v4-7)
//                 lanes16-31: M=lane-16, K = k0+8..15, k0+24..31
// ---------------------------------------------------------------------------
__device__ __forceinline__ v16bf load_frag_a(const bf16* __restrict__ A, int lda,
                                             int m0, int k0) {
  const int lane = threadIdx.x & 31;
  const int half = lane >> 4;
  const bf16* p = A + (size_t)(m0 + (lane & 15)) * lda + k0 + half * 8;
  v8bf lo = *(const v8bf*)p;
  v8bf hi = *(const v8bf*)(p + 16);
  v16bf r;
#pragma unroll
  for (int i = 0; i < 8; ++i) { r[i] = lo[i]; r[i + 8] = hi[i]; }
  return r;
}

// B: 32x16 (KxN), sourced from row-major W[N,K] (we compute A @ W^T).
// lanes 0-15: N=lane, K = k0+0..7, k0+16..23 ; lanes16-31: N=lane-16, K = k0+8..15, k0+24..31
__device__ __forceinline__ v16bf load_frag_b(const bf16* __restrict__ W, int ldb,
                                             int n0, int k0) {
  const int lane = threadIdx.x & 31;
  const int half = lane >> 4;
  const bf16* p = W + (size_t)(n0 + (lane & 15)) * ldb + k0 + half * 8;
  v8bf lo = *(const v8bf*)p;
  v8bf hi = *(const v8bf*)(p + 16);
  v16bf r;
#pragma unroll
  for (int i = 0; i < 8; ++i) { r[i] = lo[i]; r[i + 8] = hi[i]; }
  return r;
}

// ---------------------------------------------------------------------------
// Generic bf16 WMMA GEMM: C[M,N] = act(A[M,K] @ W[N,K]^T + bias)
// 8 waves / block in a 2(M) x 4(N) grid; each wave does a 32x32 tile.
// ACT: 0 = none, 1 = SiLU.  OUT_BF16: store bf16 instead of f32.
// ---------------------------------------------------------------------------
template <int ACT, bool OUT_BF16>
__global__ __launch_bounds__(256) void gemm_bf16_wmma(
    const bf16* __restrict__ A, const bf16* __restrict__ W,
    const float* __restrict__ bias, void* __restrict__ Cout,
    int M, int N, int Kd) {
  const int wave = threadIdx.x >> 5;
  const int lane = threadIdx.x & 31;
  const int m0 = blockIdx.y * 64 + (wave & 1) * 32;
  const int n0 = blockIdx.x * 128 + (wave >> 1) * 32;

  v8f acc00 = {0.f,0.f,0.f,0.f,0.f,0.f,0.f,0.f};
  v8f acc01 = acc00, acc10 = acc00, acc11 = acc00;

  for (int k = 0; k < Kd; k += 32) {
    v16bf a0 = load_frag_a(A, Kd, m0,      k);
    v16bf a1 = load_frag_a(A, Kd, m0 + 16, k);
    v16bf b0 = load_frag_b(W, Kd, n0,      k);
    v16bf b1 = load_frag_b(W, Kd, n0 + 16, k);
    acc00 = __builtin_amdgcn_wmma_f32_16x16x32_bf16(false, a0, false, b0, (short)0, acc00, false, false);
    acc01 = __builtin_amdgcn_wmma_f32_16x16x32_bf16(false, a0, false, b1, (short)0, acc01, false, false);
    acc10 = __builtin_amdgcn_wmma_f32_16x16x32_bf16(false, a1, false, b0, (short)0, acc10, false, false);
    acc11 = __builtin_amdgcn_wmma_f32_16x16x32_bf16(false, a1, false, b1, (short)0, acc11, false, false);
  }

  const int rofs = (lane >> 4) ? 8 : 0;   // C/D layout: lanes16-31 hold M = r+8
  const int cbase = lane & 15;
#pragma unroll
  for (int tm = 0; tm < 2; ++tm) {
#pragma unroll
    for (int tn = 0; tn < 2; ++tn) {
      const v8f acc = (tm == 0) ? (tn == 0 ? acc00 : acc01)
                                : (tn == 0 ? acc10 : acc11);
      const int col = n0 + tn * 16 + cbase;
      const float bv = bias ? bias[col] : 0.f;
#pragma unroll
      for (int r = 0; r < 8; ++r) {
        const int row = m0 + tm * 16 + r + rofs;
        float v = acc[r] + bv;
        if (ACT == 1) v = v / (1.f + __expf(-v));   // SiLU
        if (OUT_BF16)
          ((bf16*)Cout)[(size_t)row * N + col] = (bf16)v;
        else
          ((float*)Cout)[(size_t)row * N + col] = v;
      }
    }
  }
}

// ---------------------------------------------------------------------------
// f32 -> bf16 conversion (grid-stride)
// ---------------------------------------------------------------------------
__global__ void cvt_bf16(const float* __restrict__ s, bf16* __restrict__ d, size_t n) {
  size_t i = (size_t)blockIdx.x * blockDim.x + threadIdx.x;
  const size_t stride = (size_t)gridDim.x * blockDim.x;
  for (; i < n; i += stride) d[i] = (bf16)s[i];
}

// gy = bf16(g * y)
__global__ void mul_to_bf16(const float* __restrict__ g, const float* __restrict__ y,
                            bf16* __restrict__ d, size_t n) {
  size_t i = (size_t)blockIdx.x * blockDim.x + threadIdx.x;
  const size_t stride = (size_t)gridDim.x * blockDim.x;
  for (; i < n; i += stride) d[i] = (bf16)(g[i] * y[i]);
}

// ---------------------------------------------------------------------------
// Row RMS-norm (row length 256) -> bf16. One wave per row, 8 rows per block.
// ---------------------------------------------------------------------------
__global__ __launch_bounds__(256) void rms256_bf16(const float* __restrict__ src,
                                                   bf16* __restrict__ dst, int nrows) {
  const int wave = threadIdx.x >> 5, lane = threadIdx.x & 31;
  const int row = blockIdx.x * 8 + wave;
  if (row >= nrows) return;
  const float* r = src + (size_t)row * 256;
  float v[8]; float ss = 0.f;
#pragma unroll
  for (int j = 0; j < 8; ++j) { v[j] = r[lane + 32 * j]; ss += v[j] * v[j]; }
#pragma unroll
  for (int off = 16; off > 0; off >>= 1) ss += __shfl_xor(ss, off, 32);
  const float sc = rsqrtf(ss * (1.f / 256.f) + 1e-6f);
  bf16* d = dst + (size_t)row * 256;
#pragma unroll
  for (int j = 0; j < 8; ++j) d[lane + 32 * j] = (bf16)(v[j] * sc);
}

// ---------------------------------------------------------------------------
// Per-row top-32 of 512 scores. One wave per row (scores register-resident),
// 32 iterations of wave-argmax via shfl_xor.
// ---------------------------------------------------------------------------
__global__ __launch_bounds__(256) void topk512(const float* __restrict__ scores,
                                               float* __restrict__ s_out,
                                               int* __restrict__ i_out, int nrows) {
  const int wave = threadIdx.x >> 5, lane = threadIdx.x & 31;
  const int row = blockIdx.x * 8 + wave;
  if (row >= nrows) return;
  const float* sr = scores + (size_t)row * 512;
  float sv[16];
#pragma unroll
  for (int j = 0; j < 16; ++j) sv[j] = sr[lane + 32 * j];
  float resv = 0.f; int resi = 0;
  for (int it = 0; it < 32; ++it) {
    float bv = NEG_INF; int bi = 0x7fffffff;
#pragma unroll
    for (int j = 0; j < 16; ++j) {
      const int idx = lane + 32 * j;
      if (sv[j] > bv) { bv = sv[j]; bi = idx; }
    }
#pragma unroll
    for (int off = 16; off > 0; off >>= 1) {
      const float ov = __shfl_xor(bv, off, 32);
      const int   oi = __shfl_xor(bi, off, 32);
      if (ov > bv || (ov == bv && oi < bi)) { bv = ov; bi = oi; }
    }
    if ((bi & 31) == lane) sv[bi >> 5] = NEG_INF;   // owner invalidates
    if (lane == it) { resv = bv; resi = bi; }
  }
  s_out[(size_t)row * 32 + lane] = resv;
  i_out[(size_t)row * 32 + lane] = resi;
}

// ---------------------------------------------------------------------------
// Cartesian combine (32x32 sums), top-32, softmax, slot computation.
// One wave per row; lane a owns s1[a], scans b via shuffle.
// ---------------------------------------------------------------------------
__global__ __launch_bounds__(256) void combine_topk(
    const float* __restrict__ s1, const int* __restrict__ i1,
    const float* __restrict__ s2, const int* __restrict__ i2,
    float* __restrict__ w_out, int* __restrict__ slot_out, int nrows) {
  const int wave = threadIdx.x >> 5, lane = threadIdx.x & 31;
  const int row = blockIdx.x * 8 + wave;
  if (row >= nrows) return;
  const float a  = s1[(size_t)row * 32 + lane];
  const float sb = s2[(size_t)row * 32 + lane];
  unsigned taken = 0u;
  float selv = 0.f; int selc = 0;
  for (int it = 0; it < 32; ++it) {
    float bv = NEG_INF; int bb = -1;
#pragma unroll
    for (int b = 0; b < 32; ++b) {
      const float s2b = __shfl(sb, b, 32);
      if (!((taken >> b) & 1u)) {
        const float c = a + s2b;
        if (c > bv) { bv = c; bb = b; }
      }
    }
    int bc = (bb >= 0) ? (lane * 32 + bb) : 0;
    if (bb < 0) bv = NEG_INF;
#pragma unroll
    for (int off = 16; off > 0; off >>= 1) {
      const float ov = __shfl_xor(bv, off, 32);
      const int   oc = __shfl_xor(bc, off, 32);
      if (ov > bv || (ov == bv && oc < bc)) { bv = ov; bc = oc; }
    }
    if ((bc >> 5) == lane) taken |= 1u << (bc & 31);
    if (lane == it) { selv = bv; selc = bc; }
  }
  const int r1 = i1[(size_t)row * 32 + (selc >> 5)];
  const int r2 = i2[(size_t)row * 32 + (selc & 31)];
  // softmax across the wave's 32 selections
  float m = selv;
#pragma unroll
  for (int off = 16; off > 0; off >>= 1) { const float o = __shfl_xor(m, off, 32); m = o > m ? o : m; }
  const float e = __expf(selv - m);
  float s = e;
#pragma unroll
  for (int off = 16; off > 0; off >>= 1) s += __shfl_xor(s, off, 32);
  w_out[(size_t)row * 32 + lane]   = e / s;
  slot_out[(size_t)row * 32 + lane] = r1 * NSUB + r2;
}

// ---------------------------------------------------------------------------
// y[row,:] = sum_k w[k] * values[slot[k],:]   (512 MB gather, b128 loads)
// ---------------------------------------------------------------------------
__global__ __launch_bounds__(256) void gather_y(const float* __restrict__ values,
                                                const float* __restrict__ w,
                                                const int* __restrict__ slot,
                                                float* __restrict__ y) {
  const int row = blockIdx.x;
  __shared__ float sw[32];
  __shared__ int   ss[32];
  if (threadIdx.x < 32) {
    sw[threadIdx.x] = w[(size_t)row * 32 + threadIdx.x];
    ss[threadIdx.x] = slot[(size_t)row * 32 + threadIdx.x];
  }
  __syncthreads();
  const int col = threadIdx.x * 4;
  float4 acc = {0.f, 0.f, 0.f, 0.f};
#pragma unroll 4
  for (int k = 0; k < 32; ++k) {
    const float4 v = *(const float4*)(values + (size_t)ss[k] * DMODEL + col);
    const float wk = sw[k];
    acc.x += wk * v.x; acc.y += wk * v.y; acc.z += wk * v.z; acc.w += wk * v.w;
  }
  *(float4*)(y + (size_t)row * DMODEL + col) = acc;
}

// ---------------------------------------------------------------------------
// Host launcher
// ---------------------------------------------------------------------------
extern "C" void kernel_launch(void* const* d_in, const int* in_sizes, int n_in,
                              void* d_out, int out_size, void* d_ws, size_t ws_size,
                              hipStream_t stream) {
  const float* x    = (const float*)d_in[0];
  const float* Wq1  = (const float*)d_in[1];
  const float* bq1  = (const float*)d_in[2];
  const float* Wq2  = (const float*)d_in[3];
  const float* bq2  = (const float*)d_in[4];
  const float* sk1  = (const float*)d_in[5];
  const float* sk2  = (const float*)d_in[6];
  const float* vals = (const float*)d_in[7];
  const float* W1   = (const float*)d_in[8];
  const float* W2   = (const float*)d_in[9];
  float* out = (float*)d_out;

  char* ws = (char*)d_ws;
  auto alloc = [&](size_t bytes) -> char* {
    char* p = ws; ws += (bytes + 255) & ~(size_t)255; return p;
  };
  bf16*  x_bf    = (bf16*)alloc((size_t)NTOK * DMODEL * 2);
  bf16*  wq1_bf  = (bf16*)alloc((size_t)HDIM * DMODEL * 2);
  bf16*  wq2_bf  = (bf16*)alloc((size_t)SKDIM * HDIM * 2);
  bf16*  w1_bf   = (bf16*)alloc((size_t)DMODEL * DMODEL * 2);
  bf16*  w2_bf   = (bf16*)alloc((size_t)DMODEL * DMODEL * 2);
  bf16*  k1_bf   = (bf16*)alloc((size_t)NSUB * SKDIM * 2);
  bf16*  k2_bf   = (bf16*)alloc((size_t)NSUB * SKDIM * 2);
  bf16*  h_bf    = (bf16*)alloc((size_t)NTOK * HDIM * 2);
  float* qtmp    = (float*)alloc((size_t)NTOK * SKDIM * 4);
  bf16*  q_bf    = (bf16*)alloc((size_t)NTOK * SKDIM * 2);
  float* sc1     = (float*)alloc((size_t)NTOK * NSUB * 4);
  float* sc2     = (float*)alloc((size_t)NTOK * NSUB * 4);
  float* s1      = (float*)alloc((size_t)NTOK * TOPK * 4);
  int*   i1      = (int*)  alloc((size_t)NTOK * TOPK * 4);
  float* s2      = (float*)alloc((size_t)NTOK * TOPK * 4);
  int*   i2      = (int*)  alloc((size_t)NTOK * TOPK * 4);
  float* wsm     = (float*)alloc((size_t)NTOK * TOPK * 4);
  int*   slots   = (int*)  alloc((size_t)NTOK * TOPK * 4);
  float* ybuf    = (float*)alloc((size_t)NTOK * DMODEL * 4);
  float* gbuf    = (float*)alloc((size_t)NTOK * DMODEL * 4);
  bf16*  gy_bf   = (bf16*)alloc((size_t)NTOK * DMODEL * 2);
  (void)ws_size; (void)in_sizes; (void)n_in; (void)out_size;

  const dim3 blk(256);

  // --- precision conversion (once per call; weights fit in L2) ---
  cvt_bf16<<<2048, blk, 0, stream>>>(x,   x_bf,   (size_t)NTOK * DMODEL);
  cvt_bf16<<<2048, blk, 0, stream>>>(Wq1, wq1_bf, (size_t)HDIM * DMODEL);
  cvt_bf16<<<1024, blk, 0, stream>>>(Wq2, wq2_bf, (size_t)SKDIM * HDIM);
  cvt_bf16<<<1024, blk, 0, stream>>>(W1,  w1_bf,  (size_t)DMODEL * DMODEL);
  cvt_bf16<<<1024, blk, 0, stream>>>(W2,  w2_bf,  (size_t)DMODEL * DMODEL);
  rms256_bf16<<<NSUB / 8, blk, 0, stream>>>(sk1, k1_bf, NSUB);
  rms256_bf16<<<NSUB / 8, blk, 0, stream>>>(sk2, k2_bf, NSUB);

  // --- query MLP: h = silu(x @ Wq1^T + bq1)  [4096,4096] bf16 ---
  gemm_bf16_wmma<1, true><<<dim3(HDIM / 128, NTOK / 64), blk, 0, stream>>>(
      x_bf, wq1_bf, bq1, h_bf, NTOK, HDIM, DMODEL);
  // --- qtmp = h @ Wq2^T + bq2  [4096,256] f32, then RMS -> bf16 ---
  gemm_bf16_wmma<0, false><<<dim3(SKDIM / 128, NTOK / 64), blk, 0, stream>>>(
      h_bf, wq2_bf, bq2, qtmp, NTOK, SKDIM, HDIM);
  rms256_bf16<<<NTOK / 8, blk, 0, stream>>>(qtmp, q_bf, NTOK);

  // --- per-bank scores + top-32 ---
  gemm_bf16_wmma<0, false><<<dim3(NSUB / 128, NTOK / 64), blk, 0, stream>>>(
      q_bf, k1_bf, nullptr, sc1, NTOK, NSUB, SKDIM);
  gemm_bf16_wmma<0, false><<<dim3(NSUB / 128, NTOK / 64), blk, 0, stream>>>(
      q_bf, k2_bf, nullptr, sc2, NTOK, NSUB, SKDIM);
  topk512<<<NTOK / 8, blk, 0, stream>>>(sc1, s1, i1, NTOK);
  topk512<<<NTOK / 8, blk, 0, stream>>>(sc2, s2, i2, NTOK);

  // --- cartesian top-32 + softmax + slot ids ---
  combine_topk<<<NTOK / 8, blk, 0, stream>>>(s1, i1, s2, i2, wsm, slots, NTOK);

  // --- 512 MB value gather + weighted sum ---
  gather_y<<<NTOK, blk, 0, stream>>>(vals, wsm, slots, ybuf);

  // --- gated MLP: out = (silu(x@W1^T) * y) @ W2^T ---
  gemm_bf16_wmma<1, false><<<dim3(DMODEL / 128, NTOK / 64), blk, 0, stream>>>(
      x_bf, w1_bf, nullptr, gbuf, NTOK, DMODEL, DMODEL);
  mul_to_bf16<<<2048, blk, 0, stream>>>(gbuf, ybuf, gy_bf, (size_t)NTOK * DMODEL);
  gemm_bf16_wmma<0, false><<<dim3(DMODEL / 128, NTOK / 64), blk, 0, stream>>>(
      gy_bf, w2_bf, nullptr, out, NTOK, DMODEL, DMODEL);
}
